// BidirectionalMambaBlock_69990787055964
// MI455X (gfx1250) — matrix operand order repaired and checked by reading, not compiled, not run on hardware
//
#include <hip/hip_runtime.h>
#include <cstdint>
#include <cstddef>

// ---------------- problem constants ----------------
#define D_MODEL 1024
#define D_STATE 16
#define D_CONV  4
#define D_INNER 2048
#define DT_RANK 64
#define BATCH   4
#define SEQ     2048
#define NROWS   (BATCH * SEQ)   // 8192 token rows
#define XDBL_N  (DT_RANK + 2 * D_STATE)  // 96
#define NCHUNK  16
#define CHUNK   (SEQ / NCHUNK)  // 128

// ---------------- CDNA5 WMMA types ----------------
typedef __attribute__((ext_vector_type(16))) __bf16       v16bf;
typedef __attribute__((ext_vector_type(8)))  float        v8f;
typedef __attribute__((ext_vector_type(4)))  unsigned int u32x4;

union FragAB { v16bf v; u32x4 q[2]; };

__device__ __forceinline__ unsigned short f32_to_bf16(float f) {
  union { float f; unsigned int u; } cv; cv.f = f;
  unsigned int u = cv.u;
  unsigned int lsb = (u >> 16) & 1u;
  u += 0x7fffu + lsb;                       // round to nearest even
  return (unsigned short)(u >> 16);
}

__device__ __forceinline__ float softplus_f(float v) {
  return (v > 20.0f) ? v : log1pf(__expf(v));
}

// gfx1250 async copy: global -> LDS, 16B per lane, tracked by ASYNCcnt.
__device__ __forceinline__ void async_copy_b128(unsigned lds_off, const void* gptr) {
  asm volatile("global_load_async_to_lds_b128 %0, %1, off"
               :: "v"(lds_off), "v"((unsigned long long)(uintptr_t)gptr)
               : "memory");
}
__device__ __forceinline__ void wait_asynccnt0() {
  asm volatile("s_wait_asynccnt 0x0" ::: "memory");
}

// ---------------- fp32 -> bf16 convert ----------------
__global__ void cvt_f32_bf16(const float* __restrict__ s, unsigned short* __restrict__ d, int n) {
  int i = blockIdx.x * blockDim.x + threadIdx.x;
  if (i < n) d[i] = f32_to_bf16(s[i]);
}

// ---------------- LayerNorm -> bf16 activations ----------------
__global__ __launch_bounds__(256)
void ln_kernel(const float* __restrict__ x, const float* __restrict__ gamma,
               const float* __restrict__ beta, unsigned short* __restrict__ xn) {
  __shared__ float red[256];
  const int row = blockIdx.x;
  const int tid = threadIdx.x;
  const float* xr = x + (size_t)row * D_MODEL;
  float v[4];
  float s = 0.f;
#pragma unroll
  for (int i = 0; i < 4; ++i) { v[i] = xr[tid + 256 * i]; s += v[i]; }
  red[tid] = s; __syncthreads();
  for (int off = 128; off > 0; off >>= 1) {
    if (tid < off) red[tid] += red[tid + off];
    __syncthreads();
  }
  const float mean = red[0] * (1.0f / D_MODEL);
  __syncthreads();
  float q = 0.f;
#pragma unroll
  for (int i = 0; i < 4; ++i) { float d = v[i] - mean; q += d * d; }
  red[tid] = q; __syncthreads();
  for (int off = 128; off > 0; off >>= 1) {
    if (tid < off) red[tid] += red[tid + off];
    __syncthreads();
  }
  const float rstd = rsqrtf(red[0] * (1.0f / D_MODEL) + 1e-5f);
  unsigned short* orow = xn + (size_t)row * D_MODEL;
#pragma unroll
  for (int i = 0; i < 4; ++i) {
    const int col = tid + 256 * i;
    orow[col] = f32_to_bf16((v[i] - mean) * rstd * gamma[col] + beta[col]);
  }
}

// ---------------- WMMA bf16 GEMM: C = act(A @ W^T + bias + addsrc) ----------------
// A: MxK bf16 row-major (lda), W: NxK bf16 row-major (ldw). Requires M%128==0, K%32==0.
// Double-buffered async (global -> LDS DMA) pipeline; bf16 mirror store with its own
// stride/column-limit fuses dt extraction / bf16 re-quantization into the epilogue.
__global__ __launch_bounds__(256)
void gemm_bf16_wmma(const unsigned short* __restrict__ A, int lda,
                    const unsigned short* __restrict__ W, int ldw,
                    float* __restrict__ Cf,
                    unsigned short* __restrict__ Cb, long ldcb, int nCb,
                    const float* __restrict__ addsrc, const float* __restrict__ bias,
                    int M, int N, int K, int act) {
  __shared__ __align__(16) unsigned short sA[2][128][40];   // 32 K-halves + pad
  __shared__ __align__(16) unsigned short sB[2][128][40];

  const int tid  = threadIdx.x;
  const int lane = tid & 31;
  const int wave = tid >> 5;
  const int wm   = wave & 3;   // 4 waves along M
  const int wn   = wave >> 2;  // 2 waves along N
  const long mBlk = (long)blockIdx.y * 128;
  const long nBlk = (long)blockIdx.x * 128;

  v8f acc[2][4];
  const v8f z8 = {0.f, 0.f, 0.f, 0.f, 0.f, 0.f, 0.f, 0.f};
#pragma unroll
  for (int mt = 0; mt < 2; ++mt)
#pragma unroll
    for (int nt = 0; nt < 4; ++nt) acc[mt][nt] = z8;

  const int ldRow = tid >> 1;          // 0..127
  const int ldCol = (tid & 1) * 16;    // 0 or 16 (bf16 elements)

  // Pre-zero B rows past N in both buffers (async copies are EXEC-masked there).
  if (nBlk + 128 > (long)N) {
    const long nloc = N - nBlk;        // > 0 by grid construction
    if (ldRow >= nloc) {
      const u32x4 zz = {0, 0, 0, 0};
#pragma unroll
      for (int bufi = 0; bufi < 2; ++bufi) {
        *(u32x4*)&sB[bufi][ldRow][ldCol]     = zz;
        *(u32x4*)&sB[bufi][ldRow][ldCol + 8] = zz;
      }
    }
  }

  const int nk = K / 32;

  auto issue_tile = [&](int ki, int buf) {
    const long k0 = (long)ki * 32;
    {
      const unsigned short* gp = A + (mBlk + ldRow) * (long)lda + k0 + ldCol;
      const unsigned lo = (unsigned)(uintptr_t)&sA[buf][ldRow][ldCol];
      async_copy_b128(lo,      gp);
      async_copy_b128(lo + 16, gp + 8);
    }
    const long nrow = nBlk + ldRow;
    if (nrow < N) {
      const unsigned short* gp = W + nrow * (long)ldw + k0 + ldCol;
      const unsigned lo = (unsigned)(uintptr_t)&sB[buf][ldRow][ldCol];
      async_copy_b128(lo,      gp);
      async_copy_b128(lo + 16, gp + 8);
    }
  };

  issue_tile(0, 0);

  for (int ki = 0; ki < nk; ++ki) {
    const int cur = ki & 1;
    wait_asynccnt0();      // this wave's DMA for tile ki has landed in LDS
    __syncthreads();       // all waves' DMA landed -> tile ki published
    if (ki + 1 < nk) issue_tile(ki + 1, cur ^ 1);  // overlap next DMA w/ compute

    // A fragment (ISA 7.12.2, 16-bit A 16x32): lanes 0-15 rows M, K {0..7,16..23};
    // lanes 16-31 same rows, K {8..15,24..31}.
    FragAB aF[2];
    FragAB bF[4];
    const int rA = (lane & 15);
    const int kA = (lane >> 4) * 8;
#pragma unroll
    for (int mt = 0; mt < 2; ++mt) {
      const unsigned short* pa = &sA[cur][wm * 32 + mt * 16 + rA][kA];
      aF[mt].q[0] = *(const u32x4*)pa;
      aF[mt].q[1] = *(const u32x4*)(pa + 16);
    }
    // B fragment (32x16): lanes 0-15 col N, K {0..7,16..23}; lanes 16-31 col N, K {8..15,24..31}.
    const int kB = (lane >> 4) * 8;
#pragma unroll
    for (int nt = 0; nt < 4; ++nt) {
      const unsigned short* pb = &sB[cur][wn * 64 + nt * 16 + rA][kB];
      bF[nt].q[0] = *(const u32x4*)pb;
      bF[nt].q[1] = *(const u32x4*)(pb + 16);
    }

#pragma unroll
    for (int mt = 0; mt < 2; ++mt)
#pragma unroll
      for (int nt = 0; nt < 4; ++nt)
        acc[mt][nt] = __builtin_amdgcn_wmma_f32_16x16x32_bf16(
            false, aF[mt].v, false, bF[nt].v, (short)0, acc[mt][nt], false, false);
  }

  // C layout: VGPR r -> M = r + 8*(lane>=16); N = lane&15.
  const int colL  = (lane & 15);
  const int rowHi = (lane >> 4) * 8;
#pragma unroll
  for (int mt = 0; mt < 2; ++mt) {
#pragma unroll
    for (int nt = 0; nt < 4; ++nt) {
      const long n = nBlk + wn * 64 + nt * 16 + colL;
      if (n >= N) continue;
#pragma unroll
      for (int r = 0; r < 8; ++r) {
        const long m = mBlk + wm * 32 + mt * 16 + rowHi + r;
        if (m >= M) continue;
        float v = acc[mt][nt][r];
        if (bias)   v += bias[n];
        if (addsrc) v += addsrc[m * (long)N + n];
        if (act == 1) v = softplus_f(v);
        if (Cf) Cf[m * (long)N + n] = v;
        if (Cb && n < nCb) Cb[m * ldcb + n] = f32_to_bf16(v);
      }
    }
  }
}

// ---------------- depthwise conv (causal / anti-causal) + SiLU ----------------
__global__ void conv_silu_kernel(const float* __restrict__ xz, const float* __restrict__ cw,
                                 const float* __restrict__ cb, float* __restrict__ xc_f,
                                 unsigned short* __restrict__ xc_bf, int dir) {
  const int idx = blockIdx.x * blockDim.x + threadIdx.x;
  if (idx >= NROWS * D_INNER) return;
  const int c    = idx & (D_INNER - 1);
  const int rrow = idx >> 11;              // b*SEQ + t
  const int b    = rrow >> 11;
  const int t    = rrow & (SEQ - 1);
  float acc = cb[c];
#pragma unroll
  for (int j = 0; j < D_CONV; ++j) {
    const int tt = dir ? (t + (D_CONV - 1) - j) : (t - (D_CONV - 1) + j);
    if (tt >= 0 && tt < SEQ)
      acc += cw[c * D_CONV + j] * xz[((size_t)(b * SEQ + tt)) * (2 * D_INNER) + c];
  }
  const float s = acc / (1.f + __expf(-acc));   // SiLU
  xc_f[idx]  = s;
  xc_bf[idx] = f32_to_bf16(s);
}

// ---------------- chunked (associative) selective scan ----------------
// h_t[s] = a_t[s]*h_{t-1}[s] + b_t[s] is a diagonal linear recurrence:
// split SEQ into NCHUNK chunks; pass1 computes per-chunk (prod a, partial h),
// pass2 stitches chunk initial states, pass3 replays chunks emitting y.
struct ScanDir {
  const float* xz;
  const float* xc;
  const float* xdbl;
  const float* delta;
  const float* A_log;
  const float* Dsk;
  unsigned short* y_bf;
  float* hpart;
  float* pprod;
  float* hinit;
  int dir;
};

__global__ __launch_bounds__(256)
void scan_pass1(ScanDir a0, ScanDir a1) {
  const ScanDir a = blockIdx.y ? a1 : a0;
  const int idx   = blockIdx.x * 256 + threadIdx.x;  // (b, chunk, c)
  const int c     = idx & (D_INNER - 1);
  const int rest  = idx >> 11;
  const int chunk = rest & (NCHUNK - 1);
  const int b     = rest >> 4;
  float Av[D_STATE];
#pragma unroll
  for (int s = 0; s < D_STATE; ++s) Av[s] = -__expf(a.A_log[c * D_STATE + s]);
  float h[D_STATE], P[D_STATE];
#pragma unroll
  for (int s = 0; s < D_STATE; ++s) { h[s] = 0.f; P[s] = 1.f; }
  const int t0 = chunk * CHUNK;
  for (int j = 0; j < CHUNK; ++j) {
    const int t = a.dir ? (SEQ - 1 - (t0 + j)) : (t0 + j);
    const size_t r = (size_t)b * SEQ + t;
    const float dlt = a.delta[r * D_INNER + c];
    const float u   = a.xc[r * D_INNER + c];
    const float du  = dlt * u;
    const float* xd = a.xdbl + r * XDBL_N;
#pragma unroll
    for (int s = 0; s < D_STATE; ++s) {
      const float dA = __expf(dlt * Av[s]);
      h[s] = dA * h[s] + du * xd[DT_RANK + s];
      P[s] *= dA;
    }
  }
  const size_t base = (((size_t)b * NCHUNK + chunk) * D_INNER + c) * D_STATE;
#pragma unroll
  for (int s = 0; s < D_STATE; ++s) { a.hpart[base + s] = h[s]; a.pprod[base + s] = P[s]; }
}

__global__ __launch_bounds__(256)
void scan_pass2(ScanDir a0, ScanDir a1) {
  const ScanDir a = blockIdx.y ? a1 : a0;
  const int idx = blockIdx.x * 256 + threadIdx.x;  // (b, c)
  const int c = idx & (D_INNER - 1);
  const int b = idx >> 11;
  float h[D_STATE];
#pragma unroll
  for (int s = 0; s < D_STATE; ++s) h[s] = 0.f;
  for (int k = 0; k < NCHUNK; ++k) {
    const size_t base = (((size_t)b * NCHUNK + k) * D_INNER + c) * D_STATE;
#pragma unroll
    for (int s = 0; s < D_STATE; ++s) {
      a.hinit[base + s] = h[s];
      h[s] = a.pprod[base + s] * h[s] + a.hpart[base + s];
    }
  }
}

__global__ __launch_bounds__(256)
void scan_pass3(ScanDir a0, ScanDir a1) {
  const ScanDir a = blockIdx.y ? a1 : a0;
  const int idx   = blockIdx.x * 256 + threadIdx.x;
  const int c     = idx & (D_INNER - 1);
  const int rest  = idx >> 11;
  const int chunk = rest & (NCHUNK - 1);
  const int b     = rest >> 4;
  float Av[D_STATE];
#pragma unroll
  for (int s = 0; s < D_STATE; ++s) Av[s] = -__expf(a.A_log[c * D_STATE + s]);
  const float Dc = a.Dsk[c];
  float h[D_STATE];
  const size_t base = (((size_t)b * NCHUNK + chunk) * D_INNER + c) * D_STATE;
#pragma unroll
  for (int s = 0; s < D_STATE; ++s) h[s] = a.hinit[base + s];
  const int t0 = chunk * CHUNK;
  for (int j = 0; j < CHUNK; ++j) {
    const int t = a.dir ? (SEQ - 1 - (t0 + j)) : (t0 + j);
    const size_t r = (size_t)b * SEQ + t;
    const float dlt = a.delta[r * D_INNER + c];
    const float u   = a.xc[r * D_INNER + c];
    const float du  = dlt * u;
    const float* xd = a.xdbl + r * XDBL_N;
    float y = 0.f;
#pragma unroll
    for (int s = 0; s < D_STATE; ++s) {
      const float dA = __expf(dlt * Av[s]);
      h[s] = dA * h[s] + du * xd[DT_RANK + s];
      y += h[s] * xd[DT_RANK + D_STATE + s];
    }
    const float z = a.xz[r * (2 * D_INNER) + D_INNER + c];
    const float g = z / (1.f + __expf(-z));
    a.y_bf[r * D_INNER + c] = f32_to_bf16((y + u * Dc) * g);
  }
}

// ---------------- host orchestration ----------------
static inline void* bump(char*& p, size_t bytes) {
  void* r = p;
  p += (bytes + 255) & ~(size_t)255;
  return r;
}

extern "C" void kernel_launch(void* const* d_in, const int* in_sizes, int n_in,
                              void* d_out, int out_size, void* d_ws, size_t ws_size,
                              hipStream_t stream) {
  (void)in_sizes; (void)n_in; (void)out_size; (void)ws_size;
  const float* x      = (const float*)d_in[0];
  const float* ln_g   = (const float*)d_in[1];
  const float* ln_b   = (const float*)d_in[2];
  const float* proj_w = (const float*)d_in[21];
  const float* proj_b = (const float*)d_in[22];

  char* p = (char*)d_ws;
  unsigned short* xn_bf    = (unsigned short*)bump(p, (size_t)NROWS * D_MODEL * 2);
  unsigned short* projw_bf = (unsigned short*)bump(p, (size_t)D_MODEL * 2 * D_MODEL * 2);
  float *xz[2], *xc_f[2], *xdbl[2], *delta[2], *hpart[2], *pprod[2], *hinit[2];
  unsigned short *xc_bf[2], *dt_bf[2], *y_bf[2], *yp_bf[2];
  unsigned short *inw_bf[2], *xpw_bf[2], *dtpw_bf[2], *outw_bf[2];
  const size_t scanStateElems = (size_t)BATCH * NCHUNK * D_INNER * D_STATE;
  for (int d = 0; d < 2; ++d) {
    xz[d]      = (float*)bump(p, (size_t)NROWS * 2 * D_INNER * 4);
    xc_f[d]    = (float*)bump(p, (size_t)NROWS * D_INNER * 4);
    xc_bf[d]   = (unsigned short*)bump(p, (size_t)NROWS * D_INNER * 2);
    xdbl[d]    = (float*)bump(p, (size_t)NROWS * XDBL_N * 4);
    dt_bf[d]   = (unsigned short*)bump(p, (size_t)NROWS * DT_RANK * 2);
    delta[d]   = (float*)bump(p, (size_t)NROWS * D_INNER * 4);
    y_bf[d]    = (unsigned short*)bump(p, (size_t)NROWS * D_INNER * 2);
    yp_bf[d]   = (unsigned short*)bump(p, (size_t)NROWS * D_MODEL * 2);
    hpart[d]   = (float*)bump(p, scanStateElems * 4);
    pprod[d]   = (float*)bump(p, scanStateElems * 4);
    hinit[d]   = (float*)bump(p, scanStateElems * 4);
    inw_bf[d]  = (unsigned short*)bump(p, (size_t)2 * D_INNER * D_MODEL * 2);
    xpw_bf[d]  = (unsigned short*)bump(p, (size_t)XDBL_N * D_INNER * 2);
    dtpw_bf[d] = (unsigned short*)bump(p, (size_t)D_INNER * DT_RANK * 2);
    outw_bf[d] = (unsigned short*)bump(p, (size_t)D_MODEL * D_INNER * 2);
  }

  const dim3 blk(256);
  auto cvt = [&](const float* s, unsigned short* dst, size_t n) {
    cvt_f32_bf16<<<(unsigned)((n + 255) / 256), blk, 0, stream>>>(s, dst, (int)n);
  };

  for (int d = 0; d < 2; ++d) {
    const int base = 3 + d * 9;
    cvt((const float*)d_in[base + 0], inw_bf[d],  (size_t)2 * D_INNER * D_MODEL);
    cvt((const float*)d_in[base + 3], xpw_bf[d],  (size_t)XDBL_N * D_INNER);
    cvt((const float*)d_in[base + 4], dtpw_bf[d], (size_t)D_INNER * DT_RANK);
    cvt((const float*)d_in[base + 8], outw_bf[d], (size_t)D_MODEL * D_INNER);
  }
  cvt(proj_w, projw_bf, (size_t)D_MODEL * 2 * D_MODEL);

  ln_kernel<<<NROWS, blk, 0, stream>>>(x, ln_g, ln_b, xn_bf);

  // Pre-scan pipeline for both directions (all token-parallel, WMMA-heavy).
  for (int d = 0; d < 2; ++d) {
    const int base = 3 + d * 9;
    const float* conv_w = (const float*)d_in[base + 1];
    const float* conv_b = (const float*)d_in[base + 2];
    const float* dtp_b  = (const float*)d_in[base + 5];

    // in_proj: xz = xn @ in_w^T   (8192 x 4096, K=1024)
    gemm_bf16_wmma<<<dim3((2 * D_INNER) / 128, NROWS / 128), blk, 0, stream>>>(
        xn_bf, D_MODEL, inw_bf[d], D_MODEL, xz[d], nullptr, 0, 0, nullptr, nullptr,
        NROWS, 2 * D_INNER, D_MODEL, 0);

    conv_silu_kernel<<<(NROWS * D_INNER) / 256, blk, 0, stream>>>(
        xz[d], conv_w, conv_b, xc_f[d], xc_bf[d], d);

    // x_proj: x_dbl = xc @ xp_w^T (8192 x 96, K=2048); dt cols mirrored to bf16
    gemm_bf16_wmma<<<dim3(1, NROWS / 128), blk, 0, stream>>>(
        xc_bf[d], D_INNER, xpw_bf[d], D_INNER, xdbl[d],
        dt_bf[d], DT_RANK, DT_RANK, nullptr, nullptr,
        NROWS, XDBL_N, D_INNER, 0);

    // dt_proj + softplus: delta = softplus(dt @ dtp_w^T + dtp_b)  (8192 x 2048, K=64)
    gemm_bf16_wmma<<<dim3(D_INNER / 128, NROWS / 128), blk, 0, stream>>>(
        dt_bf[d], DT_RANK, dtpw_bf[d], DT_RANK, delta[d], nullptr, 0, 0, nullptr, dtp_b,
        NROWS, D_INNER, DT_RANK, 1);
  }

  // Chunked scan, both directions in one launch (blockIdx.y selects direction).
  ScanDir sd[2];
  for (int d = 0; d < 2; ++d) {
    const int base = 3 + d * 9;
    sd[d].xz    = xz[d];
    sd[d].xc    = xc_f[d];
    sd[d].xdbl  = xdbl[d];
    sd[d].delta = delta[d];
    sd[d].A_log = (const float*)d_in[base + 6];
    sd[d].Dsk   = (const float*)d_in[base + 7];
    sd[d].y_bf  = y_bf[d];
    sd[d].hpart = hpart[d];
    sd[d].pprod = pprod[d];
    sd[d].hinit = hinit[d];
    sd[d].dir   = d;
  }
  scan_pass1<<<dim3((BATCH * NCHUNK * D_INNER) / 256, 2), blk, 0, stream>>>(sd[0], sd[1]);
  scan_pass2<<<dim3((BATCH * D_INNER) / 256, 2), blk, 0, stream>>>(sd[0], sd[1]);
  scan_pass3<<<dim3((BATCH * NCHUNK * D_INNER) / 256, 2), blk, 0, stream>>>(sd[0], sd[1]);

  // out_proj per direction: yp = y @ out_w^T (8192 x 1024, K=2048), bf16 output
  for (int d = 0; d < 2; ++d) {
    gemm_bf16_wmma<<<dim3(D_MODEL / 128, NROWS / 128), blk, 0, stream>>>(
        y_bf[d], D_INNER, outw_bf[d], D_INNER, nullptr,
        yp_bf[d], D_MODEL, D_MODEL, nullptr, nullptr,
        NROWS, D_MODEL, D_INNER, 0);
  }

  // out = x + proj_b + yp_f @ proj_w[:, :1024]^T + yp_b @ proj_w[:, 1024:]^T
  float* out = (float*)d_out;
  gemm_bf16_wmma<<<dim3(D_MODEL / 128, NROWS / 128), blk, 0, stream>>>(
      yp_bf[0], D_MODEL, projw_bf, 2 * D_MODEL, out, nullptr, 0, 0, x, proj_b,
      NROWS, D_MODEL, D_MODEL, 0);
  gemm_bf16_wmma<<<dim3(D_MODEL / 128, NROWS / 128), blk, 0, stream>>>(
      yp_bf[1], D_MODEL, projw_bf + D_MODEL, 2 * D_MODEL, out, nullptr, 0, 0, out, nullptr,
      NROWS, D_MODEL, D_MODEL, 0);
}
